// RopeAttention_644245095155
// MI455X (gfx1250) — compile-verified, matrix operand order checked by reading, not run on hardware
//
#include <hip/hip_runtime.h>
#include <hip/hip_bf16.h>
#include <math.h>

typedef unsigned short u16;
typedef __attribute__((ext_vector_type(16))) __bf16 v16bf;
typedef __attribute__((ext_vector_type(8)))  float  v8f;
typedef __attribute__((ext_vector_type(4)))  unsigned int v4u;
typedef __attribute__((ext_vector_type(8)))  int v8i;
typedef __attribute__((ext_vector_type(4)))  int v4i;

union Frag { uint4 u[2]; v16bf v; };

__device__ __forceinline__ u16 to_bf16(float f) {
    unsigned int u = __float_as_uint(f);
    unsigned int r = u + 0x7FFFu + ((u >> 16) & 1u);
    return (u16)(r >> 16);
}

__device__ __forceinline__ v8f wmma_bf16(const Frag& a, const Frag& b, v8f c) {
    return __builtin_amdgcn_wmma_f32_16x16x32_bf16(
        false, a.v, false, b.v, (short)0, c, false, false);
}

// ---------------------------------------------------------------------------
// Tensor Data Mover: 2-D tiled global->LDS DMA (D# per CDNA5 ISA ch.8).
// pad_interval/pad_amount add LDS row padding so rows land on the same padded
// strides the compute fragments expect.
// ---------------------------------------------------------------------------
__device__ __forceinline__ void tdm_load_2d(unsigned lds_off, const void* gptr,
                                            unsigned tile_d0, unsigned tile_d1,
                                            unsigned tensor_d0, unsigned tensor_d1,
                                            unsigned stride0,
                                            unsigned pad_interval, unsigned pad_amount)
{
    unsigned long long ga = (unsigned long long)(size_t)gptr;
    v4u g0;
    g0.x = 1u;                                   // count=1 valid user descriptor
    g0.y = lds_off;                              // lds_addr
    g0.z = (unsigned)ga;                         // global_addr[31:0]
    g0.w = (unsigned)((ga >> 32) & 0x1FFFFFFu) | (2u << 30); // addr[56:32] | type=2
    v8i g1;
    g1[0] = (int)((1u << 16)                     // data_size = 1 -> 2 bytes
                | (1u << 20)                     // pad_enable
                | (pad_interval << 22)
                | (pad_amount  << 25));
    g1[1] = (int)((tensor_d0 & 0xFFFFu) << 16);                    // dim0[15:0] @bit48
    g1[2] = (int)((tensor_d0 >> 16) | ((tensor_d1 & 0xFFFFu) << 16));
    g1[3] = (int)((tensor_d1 >> 16) | (tile_d0 << 16));
    g1[4] = (int)tile_d1;                         // tile_dim1 | tile_dim2(=0)<<16
    g1[5] = (int)stride0;                         // tensor_dim0_stride[31:0]
    g1[6] = 0;                                    // stride0[47:32] | stride1 lo
    g1[7] = 0;
    v4i gz = {0, 0, 0, 0};
#if defined(__clang_major__) && (__clang_major__ >= 23)
    v8i gz8 = {0, 0, 0, 0, 0, 0, 0, 0};
    __builtin_amdgcn_tensor_load_to_lds(g0, g1, gz, gz, gz8, 0);
#else
    __builtin_amdgcn_tensor_load_to_lds(g0, g1, gz, gz, 0);
#endif
}

__device__ __forceinline__ unsigned lds_off_of(const void* p) {
    return (unsigned)(size_t)p;                  // flat LDS addr low 32 = offset
}

#define DIM_C   1024
#define SEQ_L   2048
#define NHEADS  16
#define HDIM    64
#define MROWS   8192   // B*L

// ---------------------------------------------------------------------------
// fp32 -> bf16 conversion
// ---------------------------------------------------------------------------
__global__ void cvt_f32_bf16(const float* __restrict__ src, u16* __restrict__ dst, int n) {
    int i = blockIdx.x * blockDim.x + threadIdx.x;
    if (i < n) dst[i] = to_bf16(src[i]);
}

// ---------------------------------------------------------------------------
// BF16 WMMA GEMM:  C[M,N] = A[M,K] * B[N,K]^T + bias
// A/B tiles streamed by TDM into double-buffered padded LDS tiles.
// MODE 0: fp32 out.  MODE 1: QKV epilogue (bias + RoPE -> Q/K/V [B,H,L,D] bf16)
// Block: 256 threads (8 waves). Block tile 128(M) x 64(N). Wave tile 16 x 64.
// __launch_bounds__(256,1): single resident workgroup -> big VGPR budget,
// no accumulator/fragment spills.
// ---------------------------------------------------------------------------
template <int MODE>
__global__ __launch_bounds__(256, 1)
void gemm_bf16(const u16* __restrict__ A, const u16* __restrict__ B,
               const float* __restrict__ bias,
               float* __restrict__ Cout,
               u16* __restrict__ Qo, u16* __restrict__ Ko, u16* __restrict__ Vo,
               int M, int N, int Kd)
{
    __shared__ u16 As[2][128][40];   // 32 k + 8 pad -> 80B row stride (TDM padded)
    __shared__ u16 Bs[2][64][40];

    const int tid  = threadIdx.x;
    const int w    = tid >> 5;
    const int t    = tid & 31;
    const int half = t >> 4;
    const int tn   = t & 15;

    const int m0 = blockIdx.y * 128;
    const int n0 = blockIdx.x * 64;

    v8f acc0 = {}, acc1 = {}, acc2 = {}, acc3 = {};

    // Prologue: post tile 0 (rows of 32 bf16 = 16 dwords -> interval=3, pad 4 dw -> 3)
    if (w == 0) {
        tdm_load_2d(lds_off_of(&As[0][0][0]), A + (size_t)m0 * Kd,
                    32, 128, (unsigned)Kd, (unsigned)M, (unsigned)Kd, 3, 3);
        tdm_load_2d(lds_off_of(&Bs[0][0][0]), B + (size_t)n0 * Kd,
                    32, 64, (unsigned)Kd, (unsigned)N, (unsigned)Kd, 3, 3);
    }

    for (int k0 = 0; k0 < Kd; k0 += 32) {
        const int buf = (k0 >> 5) & 1;
        __syncthreads();                 // prior reads of buf^1 complete
        if (w == 0) {
            if (k0 + 32 < Kd) {          // post next pair into alternate buffer
                tdm_load_2d(lds_off_of(&As[buf ^ 1][0][0]),
                            A + (size_t)m0 * Kd + k0 + 32,
                            32, 128, (unsigned)Kd, (unsigned)M, (unsigned)Kd, 3, 3);
                tdm_load_2d(lds_off_of(&Bs[buf ^ 1][0][0]),
                            B + (size_t)n0 * Kd + k0 + 32,
                            32, 64, (unsigned)Kd, (unsigned)N, (unsigned)Kd, 3, 3);
                __builtin_amdgcn_s_wait_tensorcnt(2);  // in-order: current pair landed
            } else {
                __builtin_amdgcn_s_wait_tensorcnt(0);
            }
        }
        __syncthreads();                 // publish current tiles

        // A fragment: lane<16 -> K{0..7,16..23}, lane>=16 -> K{8..15,24..31}
        Frag a;
        const u16* ar = &As[buf][w * 16 + tn][0];
        a.u[0] = *(const uint4*)(ar + half * 8);
        a.u[1] = *(const uint4*)(ar + half * 8 + 16);

        #pragma unroll
        for (int nt = 0; nt < 4; ++nt) {
            Frag b;
            const u16* br = &Bs[buf][nt * 16 + tn][0];
            b.u[0] = *(const uint4*)(br + half * 8);
            b.u[1] = *(const uint4*)(br + half * 8 + 16);
            switch (nt) {
                case 0: acc0 = wmma_bf16(a, b, acc0); break;
                case 1: acc1 = wmma_bf16(a, b, acc1); break;
                case 2: acc2 = wmma_bf16(a, b, acc2); break;
                case 3: acc3 = wmma_bf16(a, b, acc3); break;
            }
        }
    }

    // Epilogue. C layout: lane holds col n = tn; VGPR r holds row r + half*8.
    #pragma unroll
    for (int nt = 0; nt < 4; ++nt) {
        #pragma unroll
        for (int r = 0; r < 8; ++r) {
            float v;
            switch (nt) {
                case 0: v = acc0[r]; break;
                case 1: v = acc1[r]; break;
                case 2: v = acc2[r]; break;
                default: v = acc3[r]; break;
            }
            const int gm = m0 + w * 16 + r + half * 8;
            const int gn = n0 + nt * 16 + tn;
            v += bias[gn];
            if (MODE == 0) {
                Cout[(size_t)gm * N + gn] = v;
            } else {
                // QKV epilogue: gn -> (which, head, d); gm -> (b, l)
                const float partner = __shfl_xor(v, 1);   // RoPE pair element
                const int which = gn >> 10;
                const int h     = (gn >> 6) & (NHEADS - 1);
                const int d     = gn & (HDIM - 1);
                const int bb    = gm >> 11;
                const int l     = gm & (SEQ_L - 1);
                const size_t o  = (((size_t)(bb * NHEADS + h)) * SEQ_L + l) * HDIM + d;
                u16 ob;
                if (which == 2) {
                    ob = to_bf16(v);
                } else {
                    const float ang = (float)l * powf(10000.0f, -(float)(d & 31) / 32.0f);
                    const float c = cosf(ang), s = sinf(ang);
                    const float res = (d & 1) ? (v * c + partner * s)
                                              : (v * c - partner * s);
                    ob = to_bf16(res);
                }
                if (which == 0)      Qo[o] = ob;
                else if (which == 1) Ko[o] = ob;
                else                 Vo[o] = ob;
            }
        }
    }
}

// ---------------------------------------------------------------------------
// Flash attention (online softmax). Grid: (L/128, B*H). Block 256 = 8 waves.
// K tiles streamed by TDM (double-buffered); V transposed cooperatively.
// __launch_bounds__(256,1) keeps Q fragments + accumulators in VGPRs (no spill).
// ---------------------------------------------------------------------------
__global__ __launch_bounds__(256, 1)
void flash_attn(const u16* __restrict__ Q, const u16* __restrict__ K,
                const u16* __restrict__ V, u16* __restrict__ O)
{
    __shared__ u16 Ks[2][64][72];     // [key][d], row 128B + 16B pad (TDM padded)
    __shared__ u16 Vt[64][72];        // [d][key] transposed
    __shared__ u16 Ps[8][16][72];     // per-wave P tile [m][key]

    const int tid  = threadIdx.x;
    const int w    = tid >> 5;
    const int t    = tid & 31;
    const int half = t >> 4;
    const int tn   = t & 15;

    const int bh = blockIdx.y;
    const int bb = bh >> 4;
    const int h  = bh & (NHEADS - 1);
    const u16* Qb = Q + (size_t)bh * SEQ_L * HDIM;
    const u16* Kb = K + (size_t)bh * SEQ_L * HDIM;
    const u16* Vb = V + (size_t)bh * SEQ_L * HDIM;

    const int q0 = blockIdx.x * 128 + w * 16;

    // Q fragments (resident for whole kernel): rows q0+tn, d 0..31 and 32..63
    Frag qa0, qa1;
    {
        const u16* qrow = Qb + (size_t)(q0 + tn) * HDIM;
        qa0.u[0] = *(const uint4*)(qrow + half * 8);
        qa0.u[1] = *(const uint4*)(qrow + half * 8 + 16);
        qa1.u[0] = *(const uint4*)(qrow + 32 + half * 8);
        qa1.u[1] = *(const uint4*)(qrow + 32 + half * 8 + 16);
    }

    v8f o0 = {}, o1 = {}, o2 = {}, o3 = {};
    float mrun[8], lrun[8];
    #pragma unroll
    for (int r = 0; r < 8; ++r) { mrun[r] = -1e30f; lrun[r] = 0.0f; }

    const float sc = 0.125f; // 1/sqrt(64)
    const int ntiles = SEQ_L / 64;

    // Prologue: post K tile 0 (rows 64 bf16 = 32 dwords -> interval=4, pad 4 dw -> 3)
    if (w == 0)
        tdm_load_2d(lds_off_of(&Ks[0][0][0]), Kb,
                    64, 64, HDIM, SEQ_L, HDIM, 4, 3);

    for (int it = 0; it < ntiles; ++it) {
        const int k0  = it * 64;
        const int buf = it & 1;
        __syncthreads();               // prev compute done; Vt and Ks[buf^1] free

        // Stage V transposed: 64 keys x 64 d (2 x uint4 per thread)
        #pragma unroll
        for (int i = 0; i < 2; ++i) {
            int idx = i * 256 + tid;          // 0..511
            int key = idx >> 3, dc = idx & 7; // dc: 8-element d chunk
            uint4 vv = *(const uint4*)(Vb + (size_t)(k0 + key) * HDIM + dc * 8);
            const u16* pv = (const u16*)&vv;
            #pragma unroll
            for (int e = 0; e < 8; ++e) Vt[dc * 8 + e][key] = pv[e];
        }

        if (w == 0) {
            if (it + 1 < ntiles) {
                tdm_load_2d(lds_off_of(&Ks[buf ^ 1][0][0]),
                            Kb + (size_t)(k0 + 64) * HDIM,
                            64, 64, HDIM, SEQ_L, HDIM, 4, 3);
                __builtin_amdgcn_s_wait_tensorcnt(1);  // tile `it` landed
            } else {
                __builtin_amdgcn_s_wait_tensorcnt(0);
            }
        }
        __syncthreads();               // publish Ks[buf] + Vt

        // Scores: S(16q x 64k) = Q(16x64) x K^T, 4 tiles x 2 WMMA (d split)
        v8f s0 = {}, s1 = {}, s2 = {}, s3 = {};
        #pragma unroll
        for (int nt = 0; nt < 4; ++nt) {
            Frag b0, b1;
            const u16* krow = &Ks[buf][nt * 16 + tn][0];
            b0.u[0] = *(const uint4*)(krow + half * 8);
            b0.u[1] = *(const uint4*)(krow + half * 8 + 16);
            b1.u[0] = *(const uint4*)(krow + 32 + half * 8);
            b1.u[1] = *(const uint4*)(krow + 48 + half * 8);
            v8f c = {};
            c = wmma_bf16(qa0, b0, c);
            c = wmma_bf16(qa1, b1, c);
            switch (nt) { case 0: s0 = c; break; case 1: s1 = c; break;
                          case 2: s2 = c; break; default: s3 = c; break; }
        }

        // Online softmax per row (row m = r + half*8, spread over 16 lanes)
        #pragma unroll
        for (int r = 0; r < 8; ++r) {
            float x0 = s0[r] * sc, x1 = s1[r] * sc, x2 = s2[r] * sc, x3 = s3[r] * sc;
            float cm = fmaxf(fmaxf(x0, x1), fmaxf(x2, x3));
            #pragma unroll
            for (int off = 1; off < 16; off <<= 1)
                cm = fmaxf(cm, __shfl_xor(cm, off, 16));
            const float mnew  = fmaxf(mrun[r], cm);
            const float alpha = __expf(mrun[r] - mnew);
            const float p0 = __expf(x0 - mnew), p1 = __expf(x1 - mnew);
            const float p2 = __expf(x2 - mnew), p3 = __expf(x3 - mnew);
            float ps = p0 + p1 + p2 + p3;
            #pragma unroll
            for (int off = 1; off < 16; off <<= 1)
                ps += __shfl_xor(ps, off, 16);
            lrun[r] = lrun[r] * alpha + ps;
            mrun[r] = mnew;
            o0[r] *= alpha; o1[r] *= alpha; o2[r] *= alpha; o3[r] *= alpha;
            const int m = r + half * 8;
            Ps[w][m][     tn] = to_bf16(p0);
            Ps[w][m][16 + tn] = to_bf16(p1);
            Ps[w][m][32 + tn] = to_bf16(p2);
            Ps[w][m][48 + tn] = to_bf16(p3);
        }

        // P x V: contract over 64 keys -> 4 d-tiles x 2 WMMA
        Frag pa0, pa1;
        {
            const u16* prow = &Ps[w][tn][0];
            pa0.u[0] = *(const uint4*)(prow + half * 8);
            pa0.u[1] = *(const uint4*)(prow + half * 8 + 16);
            pa1.u[0] = *(const uint4*)(prow + 32 + half * 8);
            pa1.u[1] = *(const uint4*)(prow + 48 + half * 8);
        }
        #pragma unroll
        for (int nt = 0; nt < 4; ++nt) {
            Frag vb0, vb1;
            const u16* vrow = &Vt[nt * 16 + tn][0];
            vb0.u[0] = *(const uint4*)(vrow + half * 8);
            vb0.u[1] = *(const uint4*)(vrow + half * 8 + 16);
            vb1.u[0] = *(const uint4*)(vrow + 32 + half * 8);
            vb1.u[1] = *(const uint4*)(vrow + 48 + half * 8);
            switch (nt) {
                case 0: o0 = wmma_bf16(pa0, vb0, o0); o0 = wmma_bf16(pa1, vb1, o0); break;
                case 1: o1 = wmma_bf16(pa0, vb0, o1); o1 = wmma_bf16(pa1, vb1, o1); break;
                case 2: o2 = wmma_bf16(pa0, vb0, o2); o2 = wmma_bf16(pa1, vb1, o2); break;
                default: o3 = wmma_bf16(pa0, vb0, o3); o3 = wmma_bf16(pa1, vb1, o3); break;
            }
        }
    }

    // Normalize and store to [B, L, H*D] bf16
    #pragma unroll
    for (int nt = 0; nt < 4; ++nt) {
        #pragma unroll
        for (int r = 0; r < 8; ++r) {
            float v;
            switch (nt) { case 0: v = o0[r]; break; case 1: v = o1[r]; break;
                          case 2: v = o2[r]; break; default: v = o3[r]; break; }
            v /= lrun[r];
            const int m = r + half * 8;
            const size_t row = (size_t)bb * SEQ_L + (q0 + m);
            O[row * DIM_C + h * HDIM + nt * 16 + tn] = to_bf16(v);
        }
    }
}

// ---------------------------------------------------------------------------
extern "C" void kernel_launch(void* const* d_in, const int* in_sizes, int n_in,
                              void* d_out, int out_size, void* d_ws, size_t ws_size,
                              hipStream_t stream) {
    (void)in_sizes; (void)n_in; (void)out_size; (void)ws_size;
    const float* x      = (const float*)d_in[0];
    const float* qkv_w  = (const float*)d_in[1];
    const float* qkv_b  = (const float*)d_in[2];
    const float* proj_w = (const float*)d_in[3];
    const float* proj_b = (const float*)d_in[4];
    float* out = (float*)d_out;
    char*  ws  = (char*)d_ws;

    const size_t nX  = (size_t)MROWS * DIM_C;        // 8,388,608
    const size_t nWq = (size_t)3 * DIM_C * DIM_C;    // 3,145,728
    const size_t nWp = (size_t)DIM_C * DIM_C;        // 1,048,576
    const size_t nT  = (size_t)MROWS * DIM_C;        // Q/K/V/AO elems each

    size_t off = 0;
    auto take = [&](size_t bytes) { size_t o = off; off = (off + bytes + 255) & ~(size_t)255; return o; };
    u16* XH = (u16*)(ws + take(nX  * 2));
    u16* WQ = (u16*)(ws + take(nWq * 2));
    u16* WP = (u16*)(ws + take(nWp * 2));
    u16* Qb = (u16*)(ws + take(nT  * 2));
    u16* Kb = (u16*)(ws + take(nT  * 2));
    u16* Vb = (u16*)(ws + take(nT  * 2));
    u16* AO = (u16*)(ws + take(nT  * 2));

    cvt_f32_bf16<<<(unsigned)((nX  + 255) / 256), 256, 0, stream>>>(x,      XH, (int)nX);
    cvt_f32_bf16<<<(unsigned)((nWq + 255) / 256), 256, 0, stream>>>(qkv_w,  WQ, (int)nWq);
    cvt_f32_bf16<<<(unsigned)((nWp + 255) / 256), 256, 0, stream>>>(proj_w, WP, (int)nWp);

    // QKV GEMM (+bias, +RoPE, scatter to [B,H,L,D]): M=8192, N=3072, K=1024
    gemm_bf16<1><<<dim3(3 * DIM_C / 64, MROWS / 128), 256, 0, stream>>>(
        XH, WQ, qkv_b, nullptr, Qb, Kb, Vb, MROWS, 3 * DIM_C, DIM_C);

    // Flash attention: grid (L/128, B*H)
    flash_attn<<<dim3(SEQ_L / 128, 4 * NHEADS), 256, 0, stream>>>(Qb, Kb, Vb, AO);

    // Output projection: M=8192, N=1024, K=1024 -> fp32 out
    gemm_bf16<0><<<dim3(DIM_C / 64, MROWS / 128), 256, 0, stream>>>(
        AO, WP, proj_b, out, nullptr, nullptr, nullptr, MROWS, DIM_C, DIM_C);
}
